// EqvGNN_83614423318915
// MI455X (gfx1250) — matrix operand order repaired and checked by reading, not compiled, not run on hardware
//
#include <hip/hip_runtime.h>

#define GN 100000
#define GE 1600000
#define GD 128
#define GL 5
#define HID 256
#define EPSV 1e-5f

typedef __attribute__((ext_vector_type(16))) __bf16        v16bf;
typedef __attribute__((ext_vector_type(8)))  float         v8f;
typedef __attribute__((ext_vector_type(4)))  float         f32x4;
typedef __attribute__((ext_vector_type(4)))  unsigned int  u32x4;

union Frag16 {
  v16bf v;
  u32x4 q[2];
  unsigned short s[16];
};

__device__ __forceinline__ unsigned short f2bf(float f) {
  unsigned int u = __float_as_uint(f);
  u += 0x7fffu + ((u >> 16) & 1u);   // round-to-nearest-even
  return (unsigned short)(u >> 16);
}

// native L2 atomic f32 add, no return (STOREcnt path), device scope
__device__ __forceinline__ void atomAddF(float* p, float v) {
  asm volatile("global_atomic_add_f32 %0, %1, off scope:SCOPE_DEV"
               :: "v"(p), "v"(v) : "memory");
}

// ---- fp32 -> bf16 weight conversion (once per launch, tiny) ----
__global__ void cvt_bf16(const float* __restrict__ s, unsigned short* __restrict__ d, int n) {
  int i = blockIdx.x * blockDim.x + threadIdx.x;
  if (i < n) d[i] = f2bf(s[i]);
}

// ---- node input embedding: h[v] = xemb[x0] + xemb[120 + x1] ----
__global__ void node_embed(const int* __restrict__ x, const float* __restrict__ xemb,
                           float* __restrict__ h) {
  unsigned t = blockIdx.x * blockDim.x + threadIdx.x;
  int v = t >> 5;
  if (v >= GN) return;
  int c = (t & 31) * 4;
  int a0 = x[2 * v], a1 = x[2 * v + 1];
  f32x4 e0 = *(const f32x4*)(xemb + (size_t)a0 * GD + c);
  f32x4 e1 = *(const f32x4*)(xemb + (size_t)(120 + a1) * GD + c);
  *(f32x4*)(h + (size_t)v * GD + c) = e0 + e1;
}

// ---- agg init with self-loop: agg[v] = h[v] + etab[4] + etab[6] ----
__global__ void agg_init(const float* __restrict__ h, const float* __restrict__ etl,
                         float* __restrict__ agg) {
  unsigned t = blockIdx.x * blockDim.x + threadIdx.x;
  int v = t >> 5;
  if (v >= GN) return;
  int c = (t & 31) * 4;
  f32x4 hv = *(const f32x4*)(h + (size_t)v * GD + c);
  f32x4 e0 = *(const f32x4*)(etl + 4 * GD + c);
  f32x4 e1 = *(const f32x4*)(etl + 6 * GD + c);
  *(f32x4*)(agg + (size_t)v * GD + c) = hv + e0 + e1;
}

// ---- edge scatter: agg[dst] += h[src] + etab[bond] + etab[6+dir] ----
// one wave per edge, lane handles 4 consecutive floats (float4 gather, 4x f32 atomics)
__global__ void edge_scatter(const float* __restrict__ h, const int* __restrict__ ei,
                             const int* __restrict__ ea, const float* __restrict__ etl,
                             float* __restrict__ agg) {
  unsigned t = blockIdx.x * blockDim.x + threadIdx.x;
  int e = t >> 5;
  if (e >= GE) return;
  int c = (t & 31) * 4;
  int s  = ei[e];
  int d  = ei[GE + e];
  int bt = ea[2 * e];
  int dr = ea[2 * e + 1];
  f32x4 hv = *(const f32x4*)(h + (size_t)s * GD + c);
  f32x4 eb = *(const f32x4*)(etl + bt * GD + c);
  f32x4 ed = *(const f32x4*)(etl + (6 + dr) * GD + c);
  f32x4 m = hv + eb + ed;
  float* p = agg + (size_t)d * GD + c;
  atomAddF(p + 0, m.x);
  atomAddF(p + 1, m.y);
  atomAddF(p + 2, m.z);
  atomAddF(p + 3, m.w);
}

// ---- fused GIN MLP + BN (+ReLU): one wave per 16-node tile, WMMA bf16 ----
// Tile pairs with dual accumulator chains to fill the bf16 WMMA->WMMA hazard slots.
__global__ __launch_bounds__(128) void gin_mlp(
    const float* __restrict__ agg,
    const unsigned short* __restrict__ w1b,   // [256][128] bf16, row = hidden out
    const float* __restrict__ b1,             // [256]
    const unsigned short* __restrict__ w2b,   // [128][256] bf16, row = dim out
    const float* __restrict__ b2,             // [128]
    const float* __restrict__ gm, const float* __restrict__ bt,
    const float* __restrict__ mu, const float* __restrict__ vr,
    float* __restrict__ hout, int relu_out) {
  __shared__ unsigned short smid[4 * 16 * 264];  // per-wave hmid staging, 16x256 bf16 (stride 264)
  const int tiles = GN / 16;                     // 6250
  int wv = (blockIdx.x << 2) + (threadIdx.x >> 5);
  if (wv >= tiles) return;
  int lane = threadIdx.x & 31;
  int half = lane >> 4;    // K half select (A layout)
  int mrow = lane & 15;    // M for A rows, N for B cols / C-D cols
  unsigned short* my = smid + (threadIdx.x >> 5) * (16 * 264);

  const float* arow = agg + ((size_t)wv * 16 + mrow) * GD;

  // A fragments for GEMM1 (node rows, K=128): load + convert once.
  Frag16 afr[4];
#pragma unroll
  for (int kk = 0; kk < 4; ++kk) {
    int kA = kk * 32 + half * 16;
    int kB = kk * 32 + half * 16 + 8;
    f32x4 f0 = *(const f32x4*)(arow + kA);
    f32x4 f1 = *(const f32x4*)(arow + kA + 4);
    f32x4 f2 = *(const f32x4*)(arow + kB);
    f32x4 f3 = *(const f32x4*)(arow + kB + 4);
    afr[kk].s[0]  = f2bf(f0.x); afr[kk].s[1]  = f2bf(f0.y);
    afr[kk].s[2]  = f2bf(f0.z); afr[kk].s[3]  = f2bf(f0.w);
    afr[kk].s[4]  = f2bf(f1.x); afr[kk].s[5]  = f2bf(f1.y);
    afr[kk].s[6]  = f2bf(f1.z); afr[kk].s[7]  = f2bf(f1.w);
    afr[kk].s[8]  = f2bf(f2.x); afr[kk].s[9]  = f2bf(f2.y);
    afr[kk].s[10] = f2bf(f2.z); afr[kk].s[11] = f2bf(f2.w);
    afr[kk].s[12] = f2bf(f3.x); afr[kk].s[13] = f2bf(f3.y);
    afr[kk].s[14] = f2bf(f3.z); afr[kk].s[15] = f2bf(f3.w);
  }

  // ---- GEMM1: hmid = relu(agg @ w1^T + b1), staged to LDS as bf16 ----
#pragma unroll 1
  for (int nt = 0; nt < 16; nt += 2) {
    v8f acc0 = {0.f, 0.f, 0.f, 0.f, 0.f, 0.f, 0.f, 0.f};
    v8f acc1 = {0.f, 0.f, 0.f, 0.f, 0.f, 0.f, 0.f, 0.f};
    const unsigned short* brow0 = w1b + (size_t)(nt * 16 + mrow) * GD;
    const unsigned short* brow1 = brow0 + 16 * GD;
#pragma unroll
    for (int kk = 0; kk < 4; ++kk) {
      int kc = kk * 32 + half * 16;
      Frag16 b0, b1f;
      b0.q[0]  = *(const u32x4*)(brow0 + kc);
      b0.q[1]  = *(const u32x4*)(brow0 + kc + 8);
      b1f.q[0] = *(const u32x4*)(brow1 + kc);
      b1f.q[1] = *(const u32x4*)(brow1 + kc + 8);
      acc0 = __builtin_amdgcn_wmma_f32_16x16x32_bf16(false, afr[kk].v, false, b0.v,
                                                     (short)0, acc0, false, false);
      acc1 = __builtin_amdgcn_wmma_f32_16x16x32_bf16(false, afr[kk].v, false, b1f.v,
                                                     (short)0, acc1, false, false);
    }
    int col0 = nt * 16 + mrow;
    int col1 = col0 + 16;
    float bias0 = b1[col0];
    float bias1 = b1[col1];
#pragma unroll
    for (int r = 0; r < 8; ++r) {
      float v0 = fmaxf(acc0[r] + bias0, 0.f);
      float v1 = fmaxf(acc1[r] + bias1, 0.f);
      my[(r + 8 * half) * 264 + col0] = f2bf(v0);
      my[(r + 8 * half) * 264 + col1] = f2bf(v1);
    }
  }

  // A fragments for GEMM2 from LDS staging (dt-invariant: hoist all 8)
  const unsigned short* as = my + mrow * 264;
  Frag16 am[8];
#pragma unroll
  for (int kk = 0; kk < 8; ++kk) {
    int k0 = kk * 32;
    am[kk].q[0] = *(const u32x4*)(as + k0 + half * 16);
    am[kk].q[1] = *(const u32x4*)(as + k0 + half * 16 + 8);
  }

  // ---- GEMM2: h2 = hmid @ w2^T + b2 ; BN; optional ReLU ----
#pragma unroll 1
  for (int dt = 0; dt < 8; dt += 2) {
    v8f acc0 = {0.f, 0.f, 0.f, 0.f, 0.f, 0.f, 0.f, 0.f};
    v8f acc1 = {0.f, 0.f, 0.f, 0.f, 0.f, 0.f, 0.f, 0.f};
    const unsigned short* brow0 = w2b + (size_t)(dt * 16 + mrow) * HID;
    const unsigned short* brow1 = brow0 + 16 * HID;
#pragma unroll
    for (int kk = 0; kk < 8; ++kk) {
      int kc = kk * 32 + half * 16;
      Frag16 b0, b1f;
      b0.q[0]  = *(const u32x4*)(brow0 + kc);
      b0.q[1]  = *(const u32x4*)(brow0 + kc + 8);
      b1f.q[0] = *(const u32x4*)(brow1 + kc);
      b1f.q[1] = *(const u32x4*)(brow1 + kc + 8);
      acc0 = __builtin_amdgcn_wmma_f32_16x16x32_bf16(false, am[kk].v, false, b0.v,
                                                     (short)0, acc0, false, false);
      acc1 = __builtin_amdgcn_wmma_f32_16x16x32_bf16(false, am[kk].v, false, b1f.v,
                                                     (short)0, acc1, false, false);
    }
    int d0 = dt * 16 + mrow;
    int d1 = d0 + 16;
    float scale0 = gm[d0] * rsqrtf(vr[d0] + EPSV);
    float scale1 = gm[d1] * rsqrtf(vr[d1] + EPSV);
    float shift0 = bt[d0] - mu[d0] * scale0;
    float shift1 = bt[d1] - mu[d1] * scale1;
    float bb0 = b2[d0];
    float bb1 = b2[d1];
#pragma unroll
    for (int r = 0; r < 8; ++r) {
      float x0 = (acc0[r] + bb0) * scale0 + shift0;
      float x1 = (acc1[r] + bb1) * scale1 + shift1;
      if (relu_out) { x0 = fmaxf(x0, 0.f); x1 = fmaxf(x1, 0.f); }
      size_t rowoff = ((size_t)wv * 16 + r + 8 * half) * GD;
      hout[rowoff + d0] = x0;
      hout[rowoff + d1] = x1;
    }
  }
}

extern "C" void kernel_launch(void* const* d_in, const int* in_sizes, int n_in,
                              void* d_out, int out_size, void* d_ws, size_t ws_size,
                              hipStream_t stream) {
  (void)in_sizes; (void)n_in; (void)out_size; (void)ws_size;
  const int*   x    = (const int*)d_in[0];
  const int*   ei   = (const int*)d_in[1];
  const int*   ea   = (const int*)d_in[2];
  const float* xemb = (const float*)d_in[3];
  const float* etab = (const float*)d_in[4];
  const float* w1   = (const float*)d_in[5];
  const float* b1   = (const float*)d_in[6];
  const float* w2   = (const float*)d_in[7];
  const float* b2   = (const float*)d_in[8];
  const float* gm   = (const float*)d_in[9];
  const float* bt   = (const float*)d_in[10];
  const float* mu   = (const float*)d_in[11];
  const float* vr   = (const float*)d_in[12];

  char* ws = (char*)d_ws;
  const size_t hbytes = (size_t)GN * GD * sizeof(float);  // 51.2 MB
  float* hA  = (float*)ws;
  float* agg = (float*)(ws + hbytes);
  unsigned short* w1b = (unsigned short*)(ws + 2 * hbytes);
  unsigned short* w2b = w1b + (size_t)GL * HID * GD;
  float* hB = (float*)d_out;  // ping-pong through d_out; layer 4 lands in d_out

  const int nw = GL * HID * GD;  // 163840
  cvt_bf16<<<(nw + 255) / 256, 256, 0, stream>>>(w1, w1b, nw);
  cvt_bf16<<<(nw + 255) / 256, 256, 0, stream>>>(w2, w2b, nw);

  node_embed<<<GN * 32 / 256, 256, 0, stream>>>(x, xemb, hA);

  float* hcur = hA;
  for (int l = 0; l < GL; ++l) {
    const float* etl = etab + (size_t)l * 9 * GD;
    float* hnext = (hcur == hA) ? hB : hA;
    agg_init<<<GN * 32 / 256, 256, 0, stream>>>(hcur, etl, agg);
    edge_scatter<<<GE * 32 / 256, 256, 0, stream>>>(hcur, ei, ea, etl, agg);
    gin_mlp<<<(GN / 16 + 3) / 4, 128, 0, stream>>>(
        agg, w1b + (size_t)l * HID * GD, b1 + l * HID,
        w2b + (size_t)l * GD * HID, b2 + l * GD,
        gm + l * GD, bt + l * GD, mu + l * GD, vr + l * GD,
        hnext, (l < GL - 1) ? 1 : 0);
    hcur = hnext;
  }
  // emb->hA, l0->d_out, l1->hA, l2->d_out, l3->hA, l4->d_out  ✓
}
